// LSTM_38190849196340
// MI455X (gfx1250) — matrix operand order repaired and checked by reading, not compiled, not run on hardware
//
#include <hip/hip_runtime.h>

// ---------------------------------------------------------------------------
// LSTM forward for MI455X (gfx1250, wave32, WMMA), fused step kernel:
//   per step: acc[g|f|i|o] = [h | x_t] @ [Wh ; Wx]  (bf16 WMMA, f32 accum)
//             gates = tanh(acc + bias); c = g*i + c*f; h' = tanh(c)*o
//   out = softmax(h @ W_ph + b_p)
// ---------------------------------------------------------------------------

typedef __attribute__((ext_vector_type(16))) __bf16 v16bf;
typedef __attribute__((ext_vector_type(8)))  float  v8f;

#define LSTM_B   256
#define LSTM_D   512
#define LSTM_T   512
#define LSTM_H   1024
#define LSTM_NG  4096   // 4*H fused gate width
#define LSTM_K   1536   // H + D fused reduction
#define LSTM_CP  1024   // padded classifier width
#define LSTM_CLS 1000

__device__ __forceinline__ unsigned short f32_to_bf16(float f) {
  unsigned int u = __float_as_uint(f);
  u += 0x7FFFu + ((u >> 16) & 1u);      // round-to-nearest-even
  return (unsigned short)(u >> 16);
}

struct FragU { union { unsigned int u[8]; v16bf v; }; };

// A fragment (16x32 bf16, row-major source): lane<16 -> K {0..7, 16..23},
// lane>=16 -> K {8..15, 24..31}.  p points at (row, kb + kA).
__device__ __forceinline__ void load_frag_a(const unsigned short* p, FragU& f) {
  const uint4* q = (const uint4*)p;
  uint4 lo = q[0];        // elements +0..7
  uint4 hi = q[2];        // elements +16..23
  f.u[0]=lo.x; f.u[1]=lo.y; f.u[2]=lo.z; f.u[3]=lo.w;
  f.u[4]=hi.x; f.u[5]=hi.y; f.u[6]=hi.z; f.u[7]=hi.w;
}

// B fragment (32x16 bf16, K-major source Wt[n][k]): lane<16 -> K {0..7, 16..23},
// lane>=16 -> K {8..15, 24..31}.  p points at (col, kb + kB).
__device__ __forceinline__ void load_frag_b(const unsigned short* p, FragU& f) {
  const uint4* q = (const uint4*)p;
  uint4 lo = q[0];        // elements +0..7
  uint4 hi = q[2];        // elements +16..23
  f.u[0]=lo.x; f.u[1]=lo.y; f.u[2]=lo.z; f.u[3]=lo.w;
  f.u[4]=hi.x; f.u[5]=hi.y; f.u[6]=hi.z; f.u[7]=hi.w;
}

__device__ __forceinline__ v8f wmma_bf16(const FragU& a, const FragU& b, v8f c) {
  return __builtin_amdgcn_wmma_f32_16x16x32_bf16(
      false, a.v, false, b.v, (short)0, c, false, false);
}

// ---------------------------------------------------------------------------
// Fused LSTM step. Block: 8 waves = 2(M) x 4(J). Wave owns a 16(M) x 16(J)
// strip replicated across the 4 gates (B rows gate*H + j). Grid: (H/64, B/32).
// Two-stage software pipeline, constant trip counts, no branches in K loop.
// ---------------------------------------------------------------------------
__global__ __launch_bounds__(256) void lstm_step_kernel(
    const unsigned short* __restrict__ h_in,
    const unsigned short* __restrict__ xsl,   // xT + t*B*D
    const unsigned short* __restrict__ Wt,    // [4H][K] K-major fused weight
    const float* __restrict__ bias,           // [4H] fused (g,f,i,o)
    float* __restrict__ c,
    unsigned short* __restrict__ h_out)
{
  const int lane = threadIdx.x & 31;
  const int wid  = threadIdx.x >> 5;
  const int m0 = blockIdx.y * 32 + (wid >> 2) * 16;
  const int j0 = blockIdx.x * 64 + (wid & 3) * 16;
  const int r  = lane & 15;
  const int kA = (lane >> 4) << 3;    // 0 or 8
  const int kB = (lane >> 4) << 3;    // 0 or 8

  const unsigned short* hrow = h_in + (size_t)(m0 + r) * LSTM_H + kA;
  const unsigned short* xrow = xsl  + (size_t)(m0 + r) * LSTM_D + kA;
  const unsigned short* bG = Wt + (size_t)(0 * LSTM_H + j0 + r) * LSTM_K + kB;
  const unsigned short* bF = Wt + (size_t)(1 * LSTM_H + j0 + r) * LSTM_K + kB;
  const unsigned short* bI = Wt + (size_t)(2 * LSTM_H + j0 + r) * LSTM_K + kB;
  const unsigned short* bO = Wt + (size_t)(3 * LSTM_H + j0 + r) * LSTM_K + kB;

  v8f aG = {}, aF = {}, aI = {}, aO = {};
  FragU A0, G0, F0, I0, O0;   // stage buffer 0
  FragU A1, G1, F1, I1, O1;   // stage buffer 1

#define STEP_LOAD(Af, Gf, Ff, If, Of, arow, ak, gk)                    \
  load_frag_a((arow) + (ak), Af);                                      \
  load_frag_b(bG + (gk), Gf); load_frag_b(bF + (gk), Ff);              \
  load_frag_b(bI + (gk), If); load_frag_b(bO + (gk), Of);

#define STEP_WMMA(Af, Gf, Ff, If, Of)                                  \
  aG = wmma_bf16(Af, Gf, aG); aF = wmma_bf16(Af, Ff, aF);              \
  aI = wmma_bf16(Af, If, aI); aO = wmma_bf16(Af, Of, aO);

  // ---- h region: K = 0 .. 1023 ----
  STEP_LOAD(A0, G0, F0, I0, O0, hrow, 0, 0);
  for (int kb = 0; kb < LSTM_H - 64; kb += 64) {
    STEP_LOAD(A1, G1, F1, I1, O1, hrow, kb + 32, kb + 32);
    STEP_WMMA(A0, G0, F0, I0, O0);
    STEP_LOAD(A0, G0, F0, I0, O0, hrow, kb + 64, kb + 64);
    STEP_WMMA(A1, G1, F1, I1, O1);
  }
  // boundary: last h stage + first x stage
  STEP_LOAD(A1, G1, F1, I1, O1, hrow, LSTM_H - 32, LSTM_H - 32);
  STEP_WMMA(A0, G0, F0, I0, O0);
  STEP_LOAD(A0, G0, F0, I0, O0, xrow, 0, LSTM_H);
  STEP_WMMA(A1, G1, F1, I1, O1);
  // ---- x region: K = 1024 .. 1535 ----
  for (int kb = 0; kb < LSTM_D - 64; kb += 64) {
    STEP_LOAD(A1, G1, F1, I1, O1, xrow, kb + 32, LSTM_H + kb + 32);
    STEP_WMMA(A0, G0, F0, I0, O0);
    STEP_LOAD(A0, G0, F0, I0, O0, xrow, kb + 64, LSTM_H + kb + 64);
    STEP_WMMA(A1, G1, F1, I1, O1);
  }
  STEP_LOAD(A1, G1, F1, I1, O1, xrow, LSTM_D - 32, LSTM_K - 32);
  STEP_WMMA(A0, G0, F0, I0, O0);
  STEP_WMMA(A1, G1, F1, I1, O1);

#undef STEP_LOAD
#undef STEP_WMMA

  // Epilogue: gates in registers -> cell update -> h' (bf16) and c (f32)
  // C/D layout: VGPR i -> M = m0 + i + (lane>=16 ? 8 : 0), N = j0 + (lane&15)
  const int mb = m0 + ((lane >> 4) << 3);
  const int j  = j0 + (lane & 15);
  const float biasG = bias[j];
  const float biasF = bias[LSTM_H + j];
  const float biasI = bias[2 * LSTM_H + j];
  const float biasO = bias[3 * LSTM_H + j];
#pragma unroll
  for (int i = 0; i < 8; ++i) {
    size_t ci = (size_t)(mb + i) * LSTM_H + j;
    float g  = tanhf(aG[i] + biasG);
    float f  = tanhf(aF[i] + biasF);
    float ii = tanhf(aI[i] + biasI);
    float o  = tanhf(aO[i] + biasO);
    float cn = g * ii + c[ci] * f;
    c[ci] = cn;
    h_out[ci] = f32_to_bf16(tanhf(cn) * o);
  }
}

// ---------------------------------------------------------------------------
// Generic pipelined WMMA GEMM for the classifier: C = A @ Bt^T.
// Block tile 32(M) x 128(N), wave tile 16x32. Ktot must be a multiple of 64.
// ---------------------------------------------------------------------------
__global__ __launch_bounds__(256) void wmma_gemm_nt_kernel(
    const unsigned short* __restrict__ A, int lda,
    const unsigned short* __restrict__ Bt, int ldb,
    float* __restrict__ C, int ldc, int Ktot)
{
  const int lane = threadIdx.x & 31;
  const int wid  = threadIdx.x >> 5;
  const int m0 = blockIdx.y * 32 + (wid >> 2) * 16;
  const int n0 = blockIdx.x * 128 + (wid & 3) * 32;
  const int r  = lane & 15;
  const int kA = (lane >> 4) << 3;
  const int kB = (lane >> 4) << 3;

  const unsigned short* arow = A + (size_t)(m0 + r) * lda + kA;
  const unsigned short* b0p  = Bt + (size_t)(n0 + r) * ldb + kB;
  const unsigned short* b1p  = b0p + (size_t)16 * ldb;

  v8f acc0 = {}, acc1 = {};
  FragU A0, B00, B01, A1, B10, B11;

#define GEMM_LOAD(Af, X0, X1, k)                                       \
  load_frag_a(arow + (k), Af);                                         \
  load_frag_b(b0p + (k), X0); load_frag_b(b1p + (k), X1);

#define GEMM_WMMA(Af, X0, X1)                                          \
  acc0 = wmma_bf16(Af, X0, acc0); acc1 = wmma_bf16(Af, X1, acc1);

  GEMM_LOAD(A0, B00, B01, 0);
  for (int kb = 0; kb < Ktot - 64; kb += 64) {
    GEMM_LOAD(A1, B10, B11, kb + 32);
    GEMM_WMMA(A0, B00, B01);
    GEMM_LOAD(A0, B00, B01, kb + 64);
    GEMM_WMMA(A1, B10, B11);
  }
  GEMM_LOAD(A1, B10, B11, Ktot - 32);
  GEMM_WMMA(A0, B00, B01);
  GEMM_WMMA(A1, B10, B11);

#undef GEMM_LOAD
#undef GEMM_WMMA

  const int mb = m0 + ((lane >> 4) << 3);
  const int nb = n0 + (lane & 15);
#pragma unroll
  for (int i = 0; i < 8; ++i) {
    float* crow = C + (size_t)(mb + i) * ldc + nb;
    crow[0]  = acc0[i];
    crow[16] = acc1[i];
  }
}

// ---------------------------------------------------------------------------
// Prep kernels
// ---------------------------------------------------------------------------

// x[B][D][T] f32 -> xT[T][B][D] bf16  (t fastest on the read side)
__global__ __launch_bounds__(256) void transpose_x_kernel(
    const float* __restrict__ x, unsigned short* __restrict__ xT)
{
  size_t idx = (size_t)blockIdx.x * 256 + threadIdx.x;  // over B*D*T
  int t = (int)(idx & 511);
  size_t bd = idx >> 9;
  int d = (int)(bd & 511);
  int b = (int)(bd >> 9);
  xT[((size_t)t * LSTM_B + b) * LSTM_D + d] = f32_to_bf16(x[idx]);
}

// Fused K-major weight: Wt[n][k], n in [0,4096), k in [0,1536)
// k < 1024: Wh_gate[k][n%1024]; else Wx_gate[k-1024][n%1024]; gate = n/1024 in [g,f,i,o]
__global__ __launch_bounds__(256) void build_w_kernel(
    const float* __restrict__ Wfx, const float* __restrict__ Wfh,
    const float* __restrict__ Wgx, const float* __restrict__ Wgh,
    const float* __restrict__ Wix, const float* __restrict__ Wih,
    const float* __restrict__ Wox, const float* __restrict__ Woh,
    unsigned short* __restrict__ Wt)
{
  size_t idx = (size_t)blockIdx.x * 256 + threadIdx.x;  // over 4096*1536
  int k = (int)(idx % LSTM_K);
  int n = (int)(idx / LSTM_K);
  int gate = n >> 10;
  int col  = n & 1023;
  const float* Wh;
  const float* Wx;
  switch (gate) {
    case 0:  Wh = Wgh; Wx = Wgx; break;
    case 1:  Wh = Wfh; Wx = Wfx; break;
    case 2:  Wh = Wih; Wx = Wix; break;
    default: Wh = Woh; Wx = Wox; break;
  }
  float v = (k < LSTM_H) ? Wh[(size_t)k * LSTM_H + col]
                         : Wx[(size_t)(k - LSTM_H) * LSTM_H + col];
  Wt[idx] = f32_to_bf16(v);
}

// Classifier weight K-major + padded: Wpt[n][k], n,k in [0,1024)
__global__ __launch_bounds__(256) void build_wp_kernel(
    const float* __restrict__ Wph, unsigned short* __restrict__ Wpt)
{
  size_t idx = (size_t)blockIdx.x * 256 + threadIdx.x;  // over 1024*1024
  int k = (int)(idx & 1023);
  int n = (int)(idx >> 10);
  float v = (n < LSTM_CLS) ? Wph[(size_t)k * LSTM_CLS + n] : 0.0f;
  Wpt[idx] = f32_to_bf16(v);
}

// Fused gate bias [g,f,i,o], padded classifier bias, zero h0 (bf16) and c (f32)
__global__ __launch_bounds__(256) void init_state_kernel(
    const float* __restrict__ bf_, const float* __restrict__ bg_,
    const float* __restrict__ bi_, const float* __restrict__ bo_,
    const float* __restrict__ bp_,
    float* __restrict__ bias, float* __restrict__ bpp,
    unsigned short* __restrict__ h0, float* __restrict__ c)
{
  int idx = blockIdx.x * 256 + threadIdx.x;  // over B*H = 262144
  if (idx < LSTM_NG) {
    int g = idx >> 10, j = idx & 1023;
    const float* src = (g == 0) ? bg_ : (g == 1) ? bf_ : (g == 2) ? bi_ : bo_;
    bias[idx] = src[j];
  }
  if (idx < LSTM_CP) bpp[idx] = (idx < LSTM_CLS) ? bp_[idx] : -1.0e30f;
  h0[idx] = 0;
  c[idx] = 0.0f;
}

// ---------------------------------------------------------------------------
// Softmax over padded logits (pad columns carry bias -1e30 -> exp == 0)
// ---------------------------------------------------------------------------
__global__ __launch_bounds__(256) void softmax_kernel(
    const float* __restrict__ logits, const float* __restrict__ bp,
    float* __restrict__ out)
{
  __shared__ float smax[8];
  __shared__ float ssum[8];
  const int m = blockIdx.x;
  const int tid = threadIdx.x;
  float v[4];
  float mx = -3.0e38f;
#pragma unroll
  for (int i = 0; i < 4; ++i) {
    int n = tid + i * 256;
    v[i] = logits[(size_t)m * LSTM_CP + n] + bp[n];
    mx = fmaxf(mx, v[i]);
  }
#pragma unroll
  for (int off = 16; off > 0; off >>= 1) mx = fmaxf(mx, __shfl_xor(mx, off, 32));
  if ((tid & 31) == 0) smax[tid >> 5] = mx;
  __syncthreads();
  float bm = smax[0];
#pragma unroll
  for (int w = 1; w < 8; ++w) bm = fmaxf(bm, smax[w]);
  float s = 0.0f;
#pragma unroll
  for (int i = 0; i < 4; ++i) { v[i] = __expf(v[i] - bm); s += v[i]; }
#pragma unroll
  for (int off = 16; off > 0; off >>= 1) s += __shfl_xor(s, off, 32);
  if ((tid & 31) == 0) ssum[tid >> 5] = s;
  __syncthreads();
  float tot = 0.0f;
#pragma unroll
  for (int w = 0; w < 8; ++w) tot += ssum[w];
  float inv = 1.0f / tot;
#pragma unroll
  for (int i = 0; i < 4; ++i) {
    int n = tid + i * 256;
    if (n < LSTM_CLS) out[(size_t)m * LSTM_CLS + n] = v[i] * inv;
  }
}

// ---------------------------------------------------------------------------
// Host launcher
// ---------------------------------------------------------------------------
extern "C" void kernel_launch(void* const* d_in, const int* in_sizes, int n_in,
                              void* d_out, int out_size, void* d_ws, size_t ws_size,
                              hipStream_t stream)
{
  (void)in_sizes; (void)n_in; (void)out_size; (void)ws_size;

  const float* x   = (const float*)d_in[0];
  const float* Wfx = (const float*)d_in[1];
  const float* Wfh = (const float*)d_in[2];
  const float* Wgx = (const float*)d_in[3];
  const float* Wgh = (const float*)d_in[4];
  const float* Wix = (const float*)d_in[5];
  const float* Wih = (const float*)d_in[6];
  const float* Wox = (const float*)d_in[7];
  const float* Woh = (const float*)d_in[8];
  const float* Wph = (const float*)d_in[9];
  const float* bf_ = (const float*)d_in[10];
  const float* bg_ = (const float*)d_in[11];
  const float* bi_ = (const float*)d_in[12];
  const float* bo_ = (const float*)d_in[13];
  const float* bp_ = (const float*)d_in[14];
  float* out = (float*)d_out;

  // Workspace carve-up (all sizes 256B-aligned already)
  char* ws = (char*)d_ws;
  size_t off = 0;
  unsigned short* xT  = (unsigned short*)(ws + off); off += (size_t)LSTM_T * LSTM_B * LSTM_D * 2;
  unsigned short* Wt  = (unsigned short*)(ws + off); off += (size_t)LSTM_NG * LSTM_K * 2;
  unsigned short* Wpt = (unsigned short*)(ws + off); off += (size_t)LSTM_CP * LSTM_H * 2;
  float* bias   = (float*)(ws + off); off += (size_t)LSTM_NG * 4;
  float* bpp    = (float*)(ws + off); off += (size_t)LSTM_CP * 4;
  unsigned short* h0 = (unsigned short*)(ws + off); off += (size_t)LSTM_B * LSTM_H * 2;
  unsigned short* h1 = (unsigned short*)(ws + off); off += (size_t)LSTM_B * LSTM_H * 2;
  float* c      = (float*)(ws + off); off += (size_t)LSTM_B * LSTM_H * 4;
  float* logits = (float*)(ws + off); off += (size_t)LSTM_B * LSTM_CP * 4;

  // ---- Prep ----
  {
    size_t n = (size_t)LSTM_B * LSTM_D * LSTM_T;
    transpose_x_kernel<<<(unsigned)(n / 256), 256, 0, stream>>>(x, xT);
  }
  {
    size_t n = (size_t)LSTM_NG * LSTM_K;
    build_w_kernel<<<(unsigned)(n / 256), 256, 0, stream>>>(
        Wfx, Wfh, Wgx, Wgh, Wix, Wih, Wox, Woh, Wt);
  }
  {
    size_t n = (size_t)LSTM_CP * LSTM_H;
    build_wp_kernel<<<(unsigned)(n / 256), 256, 0, stream>>>(Wph, Wpt);
  }
  init_state_kernel<<<(LSTM_B * LSTM_H) / 256, 256, 0, stream>>>(
      bf_, bg_, bi_, bo_, bp_, bias, bpp, h0, c);

  // ---- Recurrence: 512 sequential fused steps (h double-buffered) ----
  const dim3 step_grid(LSTM_H / 64, LSTM_B / 32);   // (16, 8) = 128 blocks
  for (int t = 0; t < LSTM_T; ++t) {
    const unsigned short* hin  = (t & 1) ? h1 : h0;
    unsigned short*       hout = (t & 1) ? h0 : h1;
    const unsigned short* xslice = xT + (size_t)t * LSTM_B * LSTM_D;
    lstm_step_kernel<<<step_grid, 256, 0, stream>>>(hin, xslice, Wt, bias, c, hout);
  }
  // T = 512 (even) -> final h lives in h0
  const unsigned short* hfin = h0;

  // ---- Classifier + softmax ----
  const dim3 proj_grid(LSTM_CP / 128, LSTM_B / 32);
  wmma_gemm_nt_kernel<<<proj_grid, 256, 0, stream>>>(
      hfin, LSTM_H, Wpt, LSTM_CP, logits, LSTM_CP, LSTM_H);
  softmax_kernel<<<LSTM_B, 256, 0, stream>>>(logits, bpp, out);
}
